// DistillationLoss_80934363726257
// MI455X (gfx1250) — compile-verified, matrix operand order checked by reading, not run on hardware
//
#include <hip/hip_runtime.h>
#include <math.h>

typedef __attribute__((ext_vector_type(16))) _Float16 v16h;
typedef __attribute__((ext_vector_type(8)))  float    v8f;

#define B_    32
#define P_    8
#define K_    17
#define H_    192
#define W_    192
#define N_POSE (K_*H_*W_)        // 626688
#define CHUNKS 12
#define CH_    (N_POSE / CHUNKS) // 52224 = 256 * 4 * 51 (exact)
#define INV_T  0.5f              // 1/TEMP, TEMP=2
#define TEMP_  2.0f
#define ALPHA_ 0.5f
#define SIGMA_ 3.0f

// ---------------- task_seg: BCEWithLogits mean ----------------
__device__ __forceinline__ float bce_term(float xv, float tv) {
    return fmaxf(xv, 0.f) - xv * tv + log1pf(expf(-fabsf(xv)));
}

__global__ void bce_kernel(const float* __restrict__ x, const float* __restrict__ t,
                           float* __restrict__ acc, int n4) {
    __shared__ float red[256];
    const float4* x4 = (const float4*)x;
    const float4* t4 = (const float4*)t;
    float s = 0.f;
    for (int i = blockIdx.x * blockDim.x + threadIdx.x; i < n4; i += gridDim.x * blockDim.x) {
        float4 xv = x4[i], tv = t4[i];
        s += bce_term(xv.x, tv.x) + bce_term(xv.y, tv.y)
           + bce_term(xv.z, tv.z) + bce_term(xv.w, tv.w);
    }
    red[threadIdx.x] = s; __syncthreads();
    for (int st = 128; st > 0; st >>= 1) {
        if (threadIdx.x < st) red[threadIdx.x] += red[threadIdx.x + st];
        __syncthreads();
    }
    if (threadIdx.x == 0) atomicAdd(acc, red[0]);
}

// ---------------- pose distill: streaming logsumexp partials ----------------
// partial tuple: {m_s, S_s, m_t, S_t, W_t}, W_t = sum w_i*(t_i-s_i)/T with w_i=exp((t_i-m_t)/T)
__device__ __forceinline__ void lse_update(float sv, float tv,
                                           float& ms, float& Ss,
                                           float& mt, float& St, float& Wt) {
    if (sv > ms) { Ss = Ss * expf((ms - sv) * INV_T) + 1.f; ms = sv; }
    else         { Ss += expf((sv - ms) * INV_T); }
    if (tv > mt) {
        float f = expf((mt - tv) * INV_T);
        St = St * f + 1.f;
        Wt = Wt * f + (tv - sv) * INV_T;
        mt = tv;
    } else {
        float w = expf((tv - mt) * INV_T);
        St += w;
        Wt += w * (tv - sv) * INV_T;
    }
}

__global__ void distill_partial_kernel(const float* __restrict__ s, const float* __restrict__ t,
                                       float* __restrict__ partials) {
    const int b = blockIdx.x / CHUNKS, c = blockIdx.x % CHUNKS;
    const int tid = threadIdx.x;
    const float4* sb4 = (const float4*)(s + (size_t)b * N_POSE + (size_t)c * CH_);
    const float4* tb4 = (const float4*)(t + (size_t)b * N_POSE + (size_t)c * CH_);
    float ms = -3.0e38f, Ss = 0.f, mt = -3.0e38f, St = 0.f, Wt = 0.f;
    const int n4 = CH_ / 4;   // 13056 = 256 * 51, exact
    for (int i = tid; i < n4; i += 256) {
        float4 sv = sb4[i], tv = tb4[i];
        lse_update(sv.x, tv.x, ms, Ss, mt, St, Wt);
        lse_update(sv.y, tv.y, ms, Ss, mt, St, Wt);
        lse_update(sv.z, tv.z, ms, Ss, mt, St, Wt);
        lse_update(sv.w, tv.w, ms, Ss, mt, St, Wt);
    }
    __shared__ float Pm[5][256];
    Pm[0][tid] = ms; Pm[1][tid] = Ss; Pm[2][tid] = mt; Pm[3][tid] = St; Pm[4][tid] = Wt;
    __syncthreads();
    for (int st = 128; st > 0; st >>= 1) {
        if (tid < st) {
            // s-part
            float m1 = Pm[0][tid], S1 = Pm[1][tid], m2 = Pm[0][tid + st], S2 = Pm[1][tid + st];
            float m = fmaxf(m1, m2);
            Pm[0][tid] = m;
            Pm[1][tid] = S1 * expf((m1 - m) * INV_T) + S2 * expf((m2 - m) * INV_T);
            // t-part with weighted sum
            float a1 = Pm[2][tid], T1 = Pm[3][tid], W1 = Pm[4][tid];
            float a2 = Pm[2][tid + st], T2 = Pm[3][tid + st], W2 = Pm[4][tid + st];
            float mm = fmaxf(a1, a2);
            float f1 = expf((a1 - mm) * INV_T), f2 = expf((a2 - mm) * INV_T);
            Pm[2][tid] = mm;
            Pm[3][tid] = T1 * f1 + T2 * f2;
            Pm[4][tid] = W1 * f1 + W2 * f2;
        }
        __syncthreads();
    }
    if (tid == 0) {
        float* o = partials + (size_t)(b * CHUNKS + c) * 5;
        o[0] = Pm[0][0]; o[1] = Pm[1][0]; o[2] = Pm[2][0]; o[3] = Pm[3][0]; o[4] = Pm[4][0];
    }
}

// ---------------- task_pose: WMMA cross-term + fused sum(pred^2) + Gram target^2 ----------------
// per-(b,k) contribution = sum(pred^2) - 2*sum_p gx_p^T Pred gy_p + sum_{pq}(gx_p.gx_q)(gy_p.gy_q)
__global__ __launch_bounds__(256)
void pose_task_kernel(const float* __restrict__ pred, const float* __restrict__ kps,
                      const int* __restrict__ vis, float* __restrict__ per) {
    const int k = blockIdx.x, b = blockIdx.y;
    const int tid = threadIdx.x;
    __shared__ float gxT[H_][16];   // [i][p]  (includes 'valid' factor; p>=8 zero)
    __shared__ float gyT[W_][16];   // [j][p]  (p>=8 zero)
    __shared__ float xs[8], ys[8], vl[8];
    __shared__ float Gx[64], Gy[64];
    __shared__ float redA[256], redB[256];
    __shared__ float sttSh;
    const float inv2s2 = 1.f / (2.f * SIGMA_ * SIGMA_);

    if (tid < 8) {
        int p = tid;
        float kx = kps[(((b * P_) + p) * K_ + k) * 2 + 0];
        float ky = kps[(((b * P_) + p) * K_ + k) * 2 + 1];
        float x = floorf(kx * (float)(W_ - 1));
        float y = floorf(ky * (float)(H_ - 1));
        int v = vis[(b * P_ + p) * K_ + k];
        float valid = (v > 0 && x >= 0.f && x < (float)W_ && y >= 0.f && y < (float)H_) ? 1.f : 0.f;
        xs[p] = x; ys[p] = y; vl[p] = valid;
    }
    __syncthreads();
    for (int e = tid; e < H_ * 16; e += 256) {
        int i = e >> 4, p = e & 15;
        float gx = 0.f, gy = 0.f;
        if (p < 8) {
            float dx = (float)i - xs[p];
            gx = vl[p] * expf(-dx * dx * inv2s2);
            float dy = (float)i - ys[p];
            gy = expf(-dy * dy * inv2s2);
        }
        gxT[i][p] = gx;
        gyT[i][p] = gy;
    }
    __syncthreads();
    // 8x8 Gram matrices for the target^2 term
    if (tid < 64) {
        int p = tid >> 3, q = tid & 7;
        float s = 0.f;
        for (int i = 0; i < H_; ++i) s += gxT[i][p] * gxT[i][q];
        Gx[tid] = s;
    } else if (tid < 128) {
        int t2 = tid - 64, p = t2 >> 3, q = t2 & 7;
        float s = 0.f;
        for (int j = 0; j < W_; ++j) s += gyT[j][p] * gyT[j][q];
        Gy[t2] = s;
    }
    __syncthreads();
    if (tid == 0) {
        float s = 0.f;
        for (int e = 0; e < 64; ++e) s += Gx[e] * Gy[e];
        sttSh = s;
    }

    // WMMA: D(16x16) strip = PredStrip(16x192) @ GY(192x16) over 6 k-steps of 32.
    const float* predbk = pred + (size_t)(b * K_ + k) * (H_ * W_);
    const int wv = tid >> 5, lane = tid & 31;
    const int mrow = lane & 15;
    const int kb = (lane >= 16) ? 8 : 0;
    float cross = 0.f, spp = 0.f;
    for (int strip = wv; strip < 12; strip += 8) {   // uniform per wave -> EXEC all 1s at WMMA
        v8f acc = {};
        const int row = strip * 16 + mrow;
        const float* rp = predbk + row * W_;
        #pragma unroll
        for (int j0 = 0; j0 < W_; j0 += 32) {
            // A fragment: rows of pred, K = j0 + kb + {0..7} and j0 + 16 + kb + {0..7}
            const float4* p0 = (const float4*)(rp + j0 + kb);
            const float4* p1 = (const float4*)(rp + j0 + kb + 16);
            float4 f0 = p0[0], f1 = p0[1], f2 = p1[0], f3 = p1[1];
            spp += f0.x*f0.x + f0.y*f0.y + f0.z*f0.z + f0.w*f0.w
                 + f1.x*f1.x + f1.y*f1.y + f1.z*f1.z + f1.w*f1.w
                 + f2.x*f2.x + f2.y*f2.y + f2.z*f2.z + f2.w*f2.w
                 + f3.x*f3.x + f3.y*f3.y + f3.z*f3.z + f3.w*f3.w;
            v16h a, bm;
            a[0]=(_Float16)f0.x; a[1]=(_Float16)f0.y; a[2]=(_Float16)f0.z; a[3]=(_Float16)f0.w;
            a[4]=(_Float16)f1.x; a[5]=(_Float16)f1.y; a[6]=(_Float16)f1.z; a[7]=(_Float16)f1.w;
            a[8]=(_Float16)f2.x; a[9]=(_Float16)f2.y; a[10]=(_Float16)f2.z; a[11]=(_Float16)f2.w;
            a[12]=(_Float16)f3.x; a[13]=(_Float16)f3.y; a[14]=(_Float16)f3.z; a[15]=(_Float16)f3.w;
            // B fragment (32x16), col N = mrow, same K mapping as A
            #pragma unroll
            for (int e = 0; e < 8; ++e) {
                bm[e]     = (_Float16)gyT[j0 + kb + e][mrow];
                bm[e + 8] = (_Float16)gyT[j0 + 16 + kb + e][mrow];
            }
            acc = __builtin_amdgcn_wmma_f32_16x16x32_f16(
                false, a, false, bm, (short)0, acc, false, false);
        }
        // D[r] = (Pred @ GY)[strip*16 + r + 8*(lane>=16)][mrow]; dot with gx column
        const int ibase = strip * 16 + ((lane >= 16) ? 8 : 0);
        #pragma unroll
        for (int r = 0; r < 8; ++r)
            cross += acc[r] * gxT[ibase + r][mrow];
    }
    redA[tid] = cross; redB[tid] = spp;
    __syncthreads();
    for (int st = 128; st > 0; st >>= 1) {
        if (tid < st) { redA[tid] += redA[tid + st]; redB[tid] += redB[tid + st]; }
        __syncthreads();
    }
    if (tid == 0) atomicAdd(&per[b], redB[0] - 2.f * redA[0] + sttSh);
}

// ---------------- final combine ----------------
// ws: [0]=bce_sum, [1..32]=per[b], [64..64+B*CHUNKS*5)=logsumexp partials
__global__ void final_kernel(const int* __restrict__ vis, const float* __restrict__ ws,
                             float* __restrict__ out) {
    __shared__ float klSh[B_], tpSh[B_];
    const int tid = threadIdx.x;
    if (tid < B_) {
        const float* part = ws + 64;
        float ms = -3.0e38f, Ss = 0.f, mt = -3.0e38f, St = 0.f, Wt = 0.f;
        for (int c = 0; c < CHUNKS; ++c) {
            const float* p = part + (size_t)(tid * CHUNKS + c) * 5;
            float m2 = p[0], S2 = p[1];
            float m = fmaxf(ms, m2);
            Ss = Ss * expf((ms - m) * INV_T) + S2 * expf((m2 - m) * INV_T); ms = m;
            float a2 = p[2], T2 = p[3], W2 = p[4];
            float mm = fmaxf(mt, a2);
            float f1 = expf((mt - mm) * INV_T), f2 = expf((a2 - mm) * INV_T);
            St = St * f1 + T2 * f2; Wt = Wt * f1 + W2 * f2; mt = mm;
        }
        float lse_s = logf(Ss) + ms * INV_T;
        float lse_t = logf(St) + mt * INV_T;
        klSh[tid] = Wt / St - lse_t + lse_s;
        float dv = 0.f;
        for (int e = 0; e < P_ * K_; ++e) dv += (float)vis[tid * P_ * K_ + e];
        tpSh[tid] = ws[1 + tid] / (dv + 1e-6f);
    }
    __syncthreads();
    if (tid == 0) {
        float kls = 0.f, tps = 0.f;
        for (int b2 = 0; b2 < B_; ++b2) { kls += klSh[b2]; tps += tpSh[b2]; }
        float pose_distill = kls / (float)B_ * (TEMP_ * TEMP_);
        float seg_distill = 0.f;  // softmax over 1-channel dim: KL is identically zero
        float task_pose = tps / (float)B_;
        float task_seg = ws[0] / (float)(B_ * H_ * W_);
        out[0] = ALPHA_ * (seg_distill + pose_distill) + (1.f - ALPHA_) * (task_seg + task_pose);
    }
}

extern "C" void kernel_launch(void* const* d_in, const int* in_sizes, int n_in,
                              void* d_out, int out_size, void* d_ws, size_t ws_size,
                              hipStream_t stream) {
    (void)in_sizes; (void)n_in; (void)out_size; (void)ws_size;
    const float* s_seg  = (const float*)d_in[0];
    const float* s_pose = (const float*)d_in[1];
    // d_in[2] = t_seg_logits: unused — 1-channel softmax KL is exactly zero.
    const float* t_pose = (const float*)d_in[3];
    const float* mask   = (const float*)d_in[4];
    const float* kps    = (const float*)d_in[5];
    const int*   vis    = (const int*)d_in[6];
    float* ws = (float*)d_ws;

    // zero accumulators (+ partial region for determinism); memset is graph-capturable
    hipMemsetAsync(d_ws, 0, (size_t)(64 + B_ * CHUNKS * 5) * sizeof(float), stream);

    bce_kernel<<<512, 256, 0, stream>>>(s_seg, mask, ws + 0, (B_ * H_ * W_) / 4);
    distill_partial_kernel<<<B_ * CHUNKS, 256, 0, stream>>>(s_pose, t_pose, ws + 64);
    dim3 grid(K_, B_);
    pose_task_kernel<<<grid, 256, 0, stream>>>(s_pose, kps, vis, ws + 1);
    final_kernel<<<1, 64, 0, stream>>>(vis, ws, (float*)d_out);
}
